// ValueNetwork1_41772851921507
// MI455X (gfx1250) — matrix operand order repaired and checked by reading, not compiled, not run on hardware
//
#include <hip/hip_runtime.h>
#include <hip/hip_bf16.h>

typedef __attribute__((ext_vector_type(16))) _Float16 v16h;
typedef __attribute__((ext_vector_type(8)))  _Float16 v8h;
typedef __attribute__((ext_vector_type(8)))  float    v8f;

#define HID   50
#define NSTEP 64
#define DVEC  13
#define GW    200     // LSTM gate width
#define GWP   208     // padded to 13*16
#define KL    64      // padded K (7 + 50 -> 64)

// ---- helpers -------------------------------------------------------------

__device__ __forceinline__ v8f wmma16(v16h a, v16h b, v8f c) {
    // (neg_a, A, neg_b, B, c_mod, C, reuse_a, reuse_b)
    return __builtin_amdgcn_wmma_f32_16x16x32_f16(false, a, false, b,
                                                  (short)0, c, false, false);
}

// A-fragment gather, contiguous K order (matches B):
// lane L: row M = L&15, Kbase = (L>=16)?16:0; half e -> K = e+Kbase
// => two contiguous 16B loads at +0 and +8 halves.
__device__ __forceinline__ v16h frag16(const _Float16* p) {
    v8h lo = *(const v8h*)p;
    v8h hi = *(const v8h*)(p + 8);
    return __builtin_shufflevector(lo, hi, 0,1,2,3,4,5,6,7,8,9,10,11,12,13,14,15);
}

__device__ __forceinline__ float sigf(float x) {
    return __builtin_amdgcn_rcpf(1.0f + __expf(-x));
}
__device__ __forceinline__ float tanhf_(float x) {
    return 2.0f * sigf(2.0f * x) - 1.0f;
}

// ---- kernel 1: LSTM over 64 steps, writes joint[B][64] = [self(6)|h(50)|0] -

__global__ __launch_bounds__(64) void lstm_kernel(
    const float* __restrict__ state,   // [8192][64][13]
    const float* __restrict__ wih,     // [7][200]
    const float* __restrict__ whh,     // [50][200]
    const float* __restrict__ bih,     // [200]
    const float* __restrict__ bhh,     // [200]
    float* __restrict__ joint)         // [8192][64]
{
    __shared__ __align__(32) _Float16 WT[GWP * KL];        // transposed [n][k]
    __shared__ __align__(32) float    bias[GWP];
    __shared__ __align__(32) _Float16 Ah[2][16 * KL];      // per-wave [r][k] = [x|h|pad]
    __shared__ __align__(32) float    gatesT[2][GWP * 16]; // per-wave [n][r]
    __shared__ __align__(32) float    cst[2][16 * HID];    // per-wave cell state

    const int tid  = threadIdx.x;
    const int lane = tid & 31;
    const int w    = tid >> 5;
    const int tileBase = blockIdx.x * 32 + w * 16;

    // stage combined transposed weights [Wih;Whh] -> WT[n][k], zero padded
    for (int idx = tid; idx < GWP * KL; idx += 64) {
        int n = idx >> 6, k = idx & 63;
        float v = 0.0f;
        if (n < GW) {
            if (k < 7)       v = wih[k * GW + n];
            else if (k < 57) v = whh[(k - 7) * GW + n];
        }
        WT[idx] = (_Float16)v;
    }
    for (int n = tid; n < GWP; n += 64)
        bias[n] = (n < GW) ? (bih[n] + bhh[n]) : 0.0f;

    _Float16* ah = Ah[w];
    float*    gT = gatesT[w];
    float*    cs = cst[w];
    for (int p = lane; p < 16 * KL; p += 32) {
        int k = p & 63;
        if (k >= 7) ah[p] = (_Float16)0.0f;   // h=0 and K padding
    }
    for (int p = lane; p < 16 * HID; p += 32) cs[p] = 0.0f;
    __syncthreads();

    const int M     = lane & 15;
    const int hiA   = (lane >> 4) << 4;   // A K-base (0 or 16)
    const int hiB   = (lane >> 4) << 4;   // B K-base (0 or 16)
    const int rbase = (lane >> 4) << 3;   // D rows (0 or 8)

#pragma unroll 1
    for (int t = 0; t < NSTEP; ++t) {
        // x_t -> Ah[r][0..6]  (16 rows * 7 = 112 items)
#pragma unroll
        for (int q = 0; q < 4; ++q) {
            int p = q * 32 + lane;
            if (p < 112) {
                int r = p / 7, k = p % 7;
                ah[r * KL + k] = (_Float16)state[(size_t)(tileBase + r) * (NSTEP * DVEC)
                                                 + t * DVEC + 6 + k];
            }
        }
        __syncthreads();

        v16h a0 = frag16(&ah[M * KL + 0  + hiA]);
        v16h a1 = frag16(&ah[M * KL + 32 + hiA]);

#pragma unroll
        for (int tt = 0; tt < 13; ++tt) {
            int n = tt * 16 + M;
            v16h b0 = *(const v16h*)&WT[n * KL + 0  + hiB];
            v16h b1 = *(const v16h*)&WT[n * KL + 32 + hiB];
            v8f acc = {};
            acc = wmma16(a0, b0, acc);
            acc = wmma16(a1, b1, acc);
            *(v8f*)&gT[n * 16 + rbase] = acc;   // transposed gate store
        }
        __syncthreads();

        // elementwise LSTM cell: 16 rows * 50 = 800 = 32 lanes * 25
#pragma unroll 1
        for (int q = 0; q < 25; ++q) {
            int p = q * 32 + lane;
            int r = p / HID, j = p % HID;
            float ig = gT[(0 * HID + j) * 16 + r] + bias[0 * HID + j];
            float fg = gT[(1 * HID + j) * 16 + r] + bias[1 * HID + j];
            float gg = gT[(2 * HID + j) * 16 + r] + bias[2 * HID + j];
            float og = gT[(3 * HID + j) * 16 + r] + bias[3 * HID + j];
            float c  = cs[r * HID + j];
            c = sigf(fg) * c + sigf(ig) * tanhf_(gg);
            cs[r * HID + j] = c;
            ah[r * KL + 7 + j] = (_Float16)(sigf(og) * tanhf_(c));
        }
        __syncthreads();
    }

    // joint[b][0..5]=self, [6..55]=h, [56..63]=0
#pragma unroll 1
    for (int q = 0; q < 32; ++q) {
        int p = q * 32 + lane;            // 16*64 items
        int r = p >> 6, cc = p & 63;
        float v;
        if (cc < 6)       v = state[(size_t)(tileBase + r) * (NSTEP * DVEC) + cc];
        else if (cc < 56) v = (float)ah[r * KL + 7 + (cc - 6)];
        else              v = 0.0f;
        joint[(size_t)(tileBase + r) * 64 + cc] = v;
    }
}

// ---- kernel 2: 4-layer MLP via WMMA --------------------------------------

// KSA: A-buffer row stride (halves); KSW: weight row stride; OS: out stride;
// KCH: K chunks of 32; NT: 16-col tiles; NV: valid output cols.
template<int KSA, int KSW, int OS, int KCH, int NT, int NV>
__device__ __forceinline__ void mlp_layer(const _Float16* inA, const _Float16* wt,
                                          _Float16* outA, const float* bias, int lane)
{
    const int M     = lane & 15;
    const int hiA   = (lane >> 4) << 4;
    const int hiB   = (lane >> 4) << 4;
    const int rbase = (lane >> 4) << 3;
    v16h afr[KCH];
#pragma unroll
    for (int c = 0; c < KCH; ++c)
        afr[c] = frag16(&inA[M * KSA + c * 32 + hiA]);
#pragma unroll
    for (int t = 0; t < NT; ++t) {
        int n = t * 16 + M;
        v8f acc = {};
#pragma unroll
        for (int c = 0; c < KCH; ++c) {
            v16h bf = *(const v16h*)&wt[n * KSW + c * 32 + hiB];
            acc = wmma16(afr[c], bf, acc);
        }
        float bb = bias[n];
#pragma unroll
        for (int v = 0; v < 8; ++v) {
            float x = acc[v] + bb;
            x = (n < NV && x > 0.0f) ? x : 0.0f;   // relu + zero pad cols
            outA[(rbase + v) * OS + n] = (_Float16)x;
        }
    }
}

__global__ __launch_bounds__(64) void mlp_kernel(
    const float* __restrict__ joint,           // [8192][64]
    const float* __restrict__ w1, const float* __restrict__ b1,  // (56,150)
    const float* __restrict__ w2, const float* __restrict__ b2,  // (150,100)
    const float* __restrict__ w3, const float* __restrict__ b3,  // (100,100)
    const float* __restrict__ w4, const float* __restrict__ b4,  // (100,1)
    float* __restrict__ out)                   // [8192]
{
    __shared__ __align__(32) _Float16 WT[112 * 160];        // shared per-layer weights
    __shared__ __align__(32) _Float16 ping[2][16 * 160];
    __shared__ __align__(32) _Float16 pong[2][16 * 160];
    __shared__ __align__(32) float bs1[160], bs2[112], bs3[112], w4s[112];
    __shared__ float b4s;

    const int tid  = threadIdx.x;
    const int lane = tid & 31;
    const int w    = tid >> 5;
    const int tileBase = blockIdx.x * 32 + w * 16;

    // stage joint -> ping (f16), zero cols 64..159 (feeds K-pad of layers 2/3)
    _Float16* pi = ping[w];
    _Float16* po = pong[w];
    for (int p = lane; p < 16 * 160; p += 32) {
        int r = p / 160, k = p % 160;
        float v = (k < 64) ? joint[(size_t)(tileBase + r) * 64 + k] : 0.0f;
        pi[p] = (_Float16)v;
    }
    for (int n = tid; n < 160; n += 64) bs1[n] = (n < 150) ? b1[n] : 0.0f;
    for (int n = tid; n < 112; n += 64) {
        bs2[n] = (n < 100) ? b2[n] : 0.0f;
        bs3[n] = (n < 100) ? b3[n] : 0.0f;
        w4s[n] = (n < 100) ? w4[n] : 0.0f;
    }
    if (tid == 0) b4s = b4[0];

    // ---- layer 1: 56(->64) x 150(->160) ----
    for (int idx = tid; idx < 160 * 64; idx += 64) {
        int n = idx >> 6, k = idx & 63;
        WT[idx] = (_Float16)((n < 150 && k < 56) ? w1[k * 150 + n] : 0.0f);
    }
    __syncthreads();
    mlp_layer<160, 64, 160, 2, 10, 150>(pi, WT, po, bs1, lane);
    __syncthreads();

    // ---- layer 2: 150(->160) x 100(->112) ----
    for (int idx = tid; idx < 112 * 160; idx += 64) {
        int n = idx / 160, k = idx % 160;
        WT[idx] = (_Float16)((n < 100 && k < 150) ? w2[k * 100 + n] : 0.0f);
    }
    __syncthreads();
    mlp_layer<160, 160, 160, 5, 7, 100>(po, WT, pi, bs2, lane);
    __syncthreads();

    // ---- layer 3: 100(->128) x 100(->112) ----
    for (int idx = tid; idx < 112 * 128; idx += 64) {
        int n = idx >> 7, k = idx & 127;
        WT[idx] = (_Float16)((n < 100 && k < 100) ? w3[k * 100 + n] : 0.0f);
    }
    __syncthreads();
    mlp_layer<160, 128, 160, 4, 7, 100>(pi, WT, po, bs3, lane);
    __syncthreads();

    // ---- layer 4: 100 -> 1 (VALU dot, 2 lanes per row) ----
    int r = lane >> 1, half = lane & 1;
    float s = 0.0f;
#pragma unroll 1
    for (int q = 0; q < 50; ++q) {
        int k = half * 50 + q;
        s += (float)po[r * 160 + k] * w4s[k];
    }
    s += __shfl_xor(s, 1, 32);
    if (half == 0) out[tileBase + r] = s + b4s;
}

// ---- host ----------------------------------------------------------------

extern "C" void kernel_launch(void* const* d_in, const int* in_sizes, int n_in,
                              void* d_out, int out_size, void* d_ws, size_t ws_size,
                              hipStream_t stream) {
    const float* state = (const float*)d_in[0];
    // indices 1..10 are the dead mlp1/attention branch (output-independent)
    const float* wih = (const float*)d_in[11];
    const float* whh = (const float*)d_in[12];
    const float* bih = (const float*)d_in[13];
    const float* bhh = (const float*)d_in[14];
    const float* w1  = (const float*)d_in[15];
    const float* b1  = (const float*)d_in[16];
    const float* w2  = (const float*)d_in[17];
    const float* b2  = (const float*)d_in[18];
    const float* w3  = (const float*)d_in[19];
    const float* b3  = (const float*)d_in[20];
    const float* w4  = (const float*)d_in[21];
    const float* b4  = (const float*)d_in[22];

    float* joint = (float*)d_ws;          // 8192 * 64 floats = 2 MB
    float* out   = (float*)d_out;

    // 8192 rows, 2 waves/WG * 16 rows/wave = 32 rows per WG
    lstm_kernel<<<256, 64, 0, stream>>>(state, wih, whh, bih, bhh, joint);
    mlp_kernel <<<256, 64, 0, stream>>>(joint, w1, b1, w2, b2, w3, b3, w4, b4, out);
}
